// VAE_Encoder_5875515261499
// MI455X (gfx1250) — compile-verified, matrix operand order checked by reading, not run on hardware
//
#include <hip/hip_runtime.h>
#include <hip/hip_bf16.h>
#include <math.h>

// ---------------------------------------------------------------------------
// VAE encoder for MI455X (gfx1250, wave32).
//  conv1 -> BN1 -> ReLU -> conv2 -> BN2 -> ReLU -> attn proj -> ReLU
//  -> 120-layer LSTM (WMMA f16, f32 accum, branch-free TRANS gate math)
//  -> per-row head (lin1/lin2/mu/lv/z)
// ---------------------------------------------------------------------------

typedef __attribute__((ext_vector_type(16))) _Float16 v16h;
typedef __attribute__((ext_vector_type(8)))  _Float16 v8h;
typedef __attribute__((ext_vector_type(8)))  float    v8f;

constexpr int kB  = 512;   // batch
constexpr int kC  = 6;     // channels
constexpr int kT  = 512;   // input length
constexpr int kH  = 12;    // hidden
constexpr int kL  = 120;   // LSTM layers
constexpr int kT1 = 508;   // after conv1 (k=5, VALID)
constexpr int kT2 = 504;   // after conv2
constexpr int kNT = kB / 16;         // 32 batch tiles of 16 rows
constexpr int kZN = kB * kL * kH;    // 737280 elems per output tensor

// Branch-free gate nonlinearities on the hardware TRANS pipes.
// exp2 saturates to +inf / 0 at the extremes, rcp(inf)=0 => correct limits
// without any clamping branches (unlike libm tanhf, which diverges).
__device__ __forceinline__ float fsigm(float x) {
    return __builtin_amdgcn_rcpf(1.f + __builtin_amdgcn_exp2f(-1.44269504088896340736f * x));
}
__device__ __forceinline__ float ftanh(float x) {
    return 1.f - 2.f * __builtin_amdgcn_rcpf(1.f + __builtin_amdgcn_exp2f(2.88539008177792681472f * x));
}

// ---------------------------------------------------------------------------
// conv1: y1[b][c][t] = bias[c] + sum_{ci,k} x[b][ci][t+k] * w[c][ci][k]
// ---------------------------------------------------------------------------
__global__ void conv1_kernel(const float* __restrict__ x, const float* __restrict__ w,
                             const float* __restrict__ bias, float* __restrict__ y) {
    int idx = blockIdx.x * blockDim.x + threadIdx.x;
    if (idx >= kB * kC * kT1) return;
    int t = idx % kT1;
    int c = (idx / kT1) % kC;
    int b = idx / (kT1 * kC);
    float acc = bias[c];
#pragma unroll
    for (int ci = 0; ci < kC; ++ci) {
        const float* xp = x + ((size_t)b * kC + ci) * kT + t;
        const float* wp = w + (c * kC + ci) * 5;
#pragma unroll
        for (int k = 0; k < 5; ++k) acc += xp[k] * wp[k];
    }
    y[idx] = acc;
}

// ---------------------------------------------------------------------------
// Training-mode BatchNorm statistics: one block per channel, deterministic
// LDS tree reduction (no atomics -> graph-capture & replay safe).
// sc[ch] = gamma*rsqrt(var+eps), sc[8+ch] = beta - mean*scale
// ---------------------------------------------------------------------------
__global__ void bnstat_kernel(const float* __restrict__ y, int Tlen,
                              const float* __restrict__ g, const float* __restrict__ be,
                              float* __restrict__ sc) {
    int ch = blockIdx.x, tid = threadIdx.x;
    __shared__ float ss[256], sq[256];
    float s = 0.f, q = 0.f;
    for (int b = 0; b < kB; ++b) {
        const float* yp = y + ((size_t)b * kC + ch) * Tlen;
        for (int t = tid; t < Tlen; t += 256) { float v = yp[t]; s += v; q += v * v; }
    }
    ss[tid] = s; sq[tid] = q;
    __syncthreads();
    for (int o = 128; o > 0; o >>= 1) {
        if (tid < o) { ss[tid] += ss[tid + o]; sq[tid] += sq[tid + o]; }
        __syncthreads();
    }
    if (tid == 0) {
        const float N = (float)(kB * Tlen);
        float mean = ss[0] / N;
        float var  = sq[0] / N - mean * mean;
        float scale = g[ch] * rsqrtf(var + 1e-5f);
        sc[ch]     = scale;
        sc[8 + ch] = be[ch] - mean * scale;
    }
}

// ---------------------------------------------------------------------------
// conv2 with BN1+ReLU applied on the fly to its inputs.
// ---------------------------------------------------------------------------
__global__ void conv2_kernel(const float* __restrict__ y1, const float* __restrict__ sc1,
                             const float* __restrict__ w, const float* __restrict__ bias,
                             float* __restrict__ y2) {
    int idx = blockIdx.x * blockDim.x + threadIdx.x;
    if (idx >= kB * kC * kT2) return;
    int t = idx % kT2;
    int c = (idx / kT2) % kC;
    int b = idx / (kT2 * kC);
    float acc = bias[c];
#pragma unroll
    for (int ci = 0; ci < kC; ++ci) {
        const float* xp = y1 + ((size_t)b * kC + ci) * kT1 + t;
        const float* wp = w + (c * kC + ci) * 5;
        float scale = sc1[ci], shift = sc1[8 + ci];
#pragma unroll
        for (int k = 0; k < 5; ++k) {
            float v = fmaxf(xp[k] * scale + shift, 0.f);
            acc += v * wp[k];
        }
    }
    y2[idx] = acc;
}

// ---------------------------------------------------------------------------
// BN2+ReLU, attention projection + ReLU, and pack into the LSTM A-fragment
// layout: xbuf[tile][t][row(16)][col(16)] f16, cols 12..15 zero-padded.
// ---------------------------------------------------------------------------
__global__ void attn_kernel(const float* __restrict__ y2, const float* __restrict__ sc2,
                            const float* __restrict__ aw, const float* __restrict__ ab,
                            _Float16* __restrict__ xbuf) {
    int idx = blockIdx.x * blockDim.x + threadIdx.x;
    if (idx >= kB * kT2) return;
    int t = idx % kT2;
    int b = idx / kT2;
    float a[kC];
#pragma unroll
    for (int ci = 0; ci < kC; ++ci)
        a[ci] = fmaxf(y2[((size_t)b * kC + ci) * kT2 + t] * sc2[ci] + sc2[8 + ci], 0.f);
    int tile = b >> 4, row = b & 15;
    _Float16* dst = xbuf + (((size_t)tile * kT2 + t) * 16 + row) * 16;
#pragma unroll
    for (int h = 0; h < kH; ++h) {
        float s = ab[h];
#pragma unroll
        for (int ci = 0; ci < kC; ++ci) s += aw[h * kC + ci] * a[ci];
        dst[h] = (_Float16)fmaxf(s, 0.f);
    }
#pragma unroll
    for (int h = kH; h < 16; ++h) dst[h] = (_Float16)0.f;
}

// ---------------------------------------------------------------------------
// Pre-swizzle LSTM weights into per-lane WMMA B-fragments (f16).
// B is K(32) x N(16):  K 0..11 = Wih cols, K 16..27 = Whh cols, rest zero.
// Lane layout mirrors the A operand: lanes<16 hold K{0..7,16..23} of col N=lane,
// lanes>=16 hold K{8..15,24..31} of col N=lane-16.
// wbuf flat index = (((l*4+n)*32 + lane)*16 + j)
// ---------------------------------------------------------------------------
__global__ void wprep_kernel(const float* __restrict__ Wih, const float* __restrict__ Whh,
                             _Float16* __restrict__ wbuf) {
    int idx = blockIdx.x * blockDim.x + threadIdx.x;
    if (idx >= kL * 4 * 32 * 16) return;
    int j    = idx & 15;
    int lane = (idx >> 4) & 31;
    int n    = (idx >> 9) & 3;
    int l    = idx >> 11;
    int col  = lane & 15;
    int base = (lane & 16) ? 8 : 0;
    int k    = (j < 8) ? (base + j) : (16 + base + (j - 8));
    float v = 0.f;
    if (col < kH) {
        int wrow = l * 48 + n * kH + col;   // gate n, output unit col
        if (k < 16) { if (k < kH) v = Wih[(size_t)wrow * kH + k]; }
        else        { int kk = k - 16; if (kk < kH) v = Whh[(size_t)wrow * kH + kk]; }
    }
    wbuf[idx] = (_Float16)v;
}

// ---------------------------------------------------------------------------
// 120-layer LSTM. One wave (32 threads) per 16-row batch tile; 32 blocks
// -> 32 WGPs each running an independent recurrence. Per step:
//   A[16x32] = [x_t | h_{t-1}]  (K split at 16 matches the A-operand halves)
//   4x v_wmma_f32_16x16x32_f16  (gate tiles i,f,g,o padded to N=16, C=bias)
// Weights stay in VGPRs for the whole layer; x_{t+1} is software-pipelined;
// h recirculates through 512B of LDS; one ds_load_b128 per step feeds both
// the next A-fragment and a single global_store_b128 to the ping buffer.
// ---------------------------------------------------------------------------
__global__ __launch_bounds__(32) void lstm_kernel(
    _Float16* __restrict__ xA, _Float16* __restrict__ xB,
    const _Float16* __restrict__ wbuf,
    const float* __restrict__ bih, const float* __restrict__ bhh,
    float* __restrict__ hall) {
    const int tile  = blockIdx.x;
    const int lane  = threadIdx.x;
    const int row   = lane & 15;          // A-operand: M row this lane loads
    const int col   = lane & 15;          // C/D layout: N column this lane owns
    const int kb    = (lane >> 4) * 8;    // A/B K-half selector
    const int rbase = (lane >> 4) * 8;    // C/D: rows r(+8) for upper lanes

    __shared__ __align__(16) _Float16 hsm[16 * 16];

    _Float16* xcur = xA + (size_t)tile * kT2 * 256;
    _Float16* xnxt = xB + (size_t)tile * kT2 * 256;
    const int roff = row * 16 + kb;       // per-lane offset inside a [16][16] slab

    for (int l = 0; l < kL; ++l) {
        // B fragments: resident in VGPRs for all 504 steps of this layer
        v16h bf[4];
#pragma unroll
        for (int n = 0; n < 4; ++n)
            bf[n] = *(const v16h*)(wbuf + (((size_t)(l * 4 + n)) * 32 + lane) * 16);
        v8f cinit[4];
#pragma unroll
        for (int n = 0; n < 4; ++n) {
            float bv = (col < kH) ? (bih[l * 48 + n * kH + col] + bhh[l * 48 + n * kH + col])
                                  : 0.f;
#pragma unroll
            for (int r = 0; r < 8; ++r) cinit[n][r] = bv;
        }

        v8f cst = {};            // c0 = 0
        v8h ha  = {};            // h0 = 0 (no LDS init needed: write-before-read)
        v8h xa  = *(const v8h*)(xcur + roff);   // x_0

        for (int t = 0; t < kT2; ++t) {
            v16h A;
#pragma unroll
            for (int j = 0; j < 8; ++j) { A[j] = xa[j]; A[8 + j] = ha[j]; }

            v8f acc[4];
#pragma unroll
            for (int n = 0; n < 4; ++n)
                acc[n] = __builtin_amdgcn_wmma_f32_16x16x32_f16(
                    false, A, false, bf[n], (short)0, cinit[n], false, false);

            // software pipeline: pull x_{t+1} while the gate math runs
            int tn = (t + 1 < kT2) ? t + 1 : t;
            xa = *(const v8h*)(xcur + (size_t)tn * 256 + roff);

            float hv[8];
#pragma unroll
            for (int r = 0; r < 8; ++r) {
                float gi = acc[0][r], gf = acc[1][r], gg = acc[2][r], go = acc[3][r];
                float c  = fsigm(gf) * cst[r] + fsigm(gi) * ftanh(gg);
                cst[r]   = c;
                hv[r]    = fsigm(go) * ftanh(c);
            }

#pragma unroll
            for (int r = 0; r < 8; ++r)
                hsm[(r + rbase) * 16 + col] = (_Float16)hv[r];   // D -> row-major
            __syncthreads();     // single-wave block: compiles to a ds wait

            ha = *(const v8h*)(hsm + roff);                       // h_t, A layout
            *(v8h*)(xnxt + (size_t)t * 256 + roff) = ha;          // next layer input

            if (t == kT2 - 1) {
#pragma unroll
                for (int r = 0; r < 8; ++r)
                    if (col < kH)
                        hall[((size_t)l * kB + tile * 16 + r + rbase) * kH + col] = hv[r];
            }
        }
        _Float16* tmp = xcur; xcur = xnxt; xnxt = tmp;
    }
}

// ---------------------------------------------------------------------------
// Head: per (b,l) row -> lin1+ReLU, lin2+ReLU, mu, log_var, z.
// Outputs concatenated flat in reference return order: z, out, mu, log_var.
// ---------------------------------------------------------------------------
__global__ void head_kernel(const float* __restrict__ hall, const float* __restrict__ eps,
                            const float* __restrict__ l1w, const float* __restrict__ l1b,
                            const float* __restrict__ l2w, const float* __restrict__ l2b,
                            const float* __restrict__ muw, const float* __restrict__ mub,
                            const float* __restrict__ lvw, const float* __restrict__ lvb,
                            float* __restrict__ out) {
    int idx = blockIdx.x * blockDim.x + threadIdx.x;
    if (idx >= kB * kL) return;
    int b = idx / kL, l = idx % kL;
    float h0[kH], h1[kH], h2[kH];
#pragma unroll
    for (int j = 0; j < kH; ++j) h0[j] = hall[((size_t)l * kB + b) * kH + j];
#pragma unroll
    for (int j = 0; j < kH; ++j) {
        float s = l1b[j];
#pragma unroll
        for (int k = 0; k < kH; ++k) s += l1w[j * kH + k] * h0[k];
        h1[j] = fmaxf(s, 0.f);
    }
#pragma unroll
    for (int j = 0; j < kH; ++j) {
        float s = l2b[j];
#pragma unroll
        for (int k = 0; k < kH; ++k) s += l2w[j * kH + k] * h1[k];
        h2[j] = fmaxf(s, 0.f);
    }
    size_t base = ((size_t)b * kL + l) * kH;
#pragma unroll
    for (int j = 0; j < kH; ++j) {
        float mu = mub[j], lv = lvb[j];
#pragma unroll
        for (int k = 0; k < kH; ++k) { mu += muw[j * kH + k] * h2[k]; lv += lvw[j * kH + k] * h2[k]; }
        float z = mu + __expf(lv * 0.5f) * eps[base + j];
        out[base + j]                   = z;      // z
        out[kZN + base + j]             = h2[j];  // out
        out[2 * (size_t)kZN + base + j] = mu;
        out[3 * (size_t)kZN + base + j] = lv;
    }
}

// ---------------------------------------------------------------------------
extern "C" void kernel_launch(void* const* d_in, const int* in_sizes, int n_in,
                              void* d_out, int out_size, void* d_ws, size_t ws_size,
                              hipStream_t stream) {
    (void)in_sizes; (void)n_in; (void)out_size; (void)ws_size;
    const float* x    = (const float*)d_in[0];
    const float* eps  = (const float*)d_in[1];
    const float* cAw  = (const float*)d_in[2];
    const float* cAb  = (const float*)d_in[3];
    const float* cGw  = (const float*)d_in[4];
    const float* cGb  = (const float*)d_in[5];
    const float* bn1g = (const float*)d_in[6];
    const float* bn1b = (const float*)d_in[7];
    const float* bn2g = (const float*)d_in[8];
    const float* bn2b = (const float*)d_in[9];
    const float* aw   = (const float*)d_in[10];
    const float* ab   = (const float*)d_in[11];
    const float* Wih  = (const float*)d_in[12];
    const float* Whh  = (const float*)d_in[13];
    const float* bih  = (const float*)d_in[14];
    const float* bhh  = (const float*)d_in[15];
    const float* l1w  = (const float*)d_in[16];
    const float* l1b  = (const float*)d_in[17];
    const float* l2w  = (const float*)d_in[18];
    const float* l2b  = (const float*)d_in[19];
    const float* muw  = (const float*)d_in[20];
    const float* mub  = (const float*)d_in[21];
    const float* lvw  = (const float*)d_in[22];
    const float* lvb  = (const float*)d_in[23];

    char* ws = (char*)d_ws;
    // Layout (bytes). xB aliases the conv buffers (dead by the time LSTM runs).
    const size_t sz_y1   = sizeof(float) * kB * kC * kT1;              // 6,242,304
    const size_t sz_y2   = sizeof(float) * kB * kC * kT2;              // 6,193,152
    const size_t sz_xbuf = sizeof(_Float16) * (size_t)kNT * kT2 * 256; // 8,257,536
    const size_t sz_w    = sizeof(_Float16) * (size_t)kL * 4 * 32 * 16;
    const size_t sz_hall = sizeof(float) * (size_t)kL * kB * kH;

    float*     y1   = (float*)(ws + 0);
    float*     y2   = (float*)(ws + sz_y1);
    _Float16*  xB   = (_Float16*)(ws + 0);                  // aliases y1/y2
    size_t off = sz_y1 + sz_y2;                             // 12,435,456
    float*     sc1  = (float*)(ws + off); off += 256;
    float*     sc2  = (float*)(ws + off); off += 256;
    _Float16*  xA   = (_Float16*)(ws + off); off += sz_xbuf;
    _Float16*  wbuf = (_Float16*)(ws + off); off += sz_w;
    float*     hall = (float*)(ws + off); off += sz_hall;   // ~24.1 MB total

    // Weight pre-swizzle is independent of the activations: launch first.
    wprep_kernel<<<(kL * 4 * 32 * 16 + 255) / 256, 256, 0, stream>>>(Wih, Whh, wbuf);
    conv1_kernel<<<(kB * kC * kT1 + 255) / 256, 256, 0, stream>>>(x, cAw, cAb, y1);
    bnstat_kernel<<<kC, 256, 0, stream>>>(y1, kT1, bn1g, bn1b, sc1);
    conv2_kernel<<<(kB * kC * kT2 + 255) / 256, 256, 0, stream>>>(y1, sc1, cGw, cGb, y2);
    bnstat_kernel<<<kC, 256, 0, stream>>>(y2, kT2, bn2g, bn2b, sc2);
    attn_kernel<<<(kB * kT2 + 255) / 256, 256, 0, stream>>>(y2, sc2, aw, ab, xA);
    lstm_kernel<<<kNT, 32, 0, stream>>>(xA, xB, wbuf, bih, bhh, hall);
    head_kernel<<<(kB * kL + 255) / 256, 256, 0, stream>>>(
        hall, eps, l1w, l1b, l2w, l2b, muw, mub, lvw, lvb, (float*)d_out);
}
